// EdgeGeneratorHeavy_57071525429608
// MI455X (gfx1250) — compile-verified, matrix operand order checked by reading, not run on hardware
//
#include <hip/hip_runtime.h>
#include <hip/hip_bf16.h>

typedef _Float16 half8  __attribute__((ext_vector_type(8)));
typedef _Float16 half16 __attribute__((ext_vector_type(16)));
typedef float    float8 __attribute__((ext_vector_type(8)));

#define F_DIM 128
#define H_DIM 64
#define WAVES_PER_BLOCK 8
#define PERSISTENT_BLOCKS 1024
#define BN_EPS 1e-5f

// Fragment counts: GEMM1: K=256 -> 8 chunks of 32, N=64 -> 4 tiles of 16.
// GEMM2: K=64 -> 2 chunks, N=64 -> 4 tiles.
#define WF1_HALFS (8 * 4 * 32 * 16)  // 16384 halfs = 32 KB
#define WF2_HALFS (2 * 4 * 32 * 16)  //  4096 halfs =  8 KB

// ---------------------------------------------------------------------------
// Prep: swizzle W_in^T / W_h^T (f32) into f16 WMMA B-fragment order in ws,
// and zero the global BN-stat accumulators.
// B-matrix (KxN, 16-bit) lane layout: lane L holds col N=(L&15); half j holds
// K = c*32 + (L>>4)*8 + (j&7) + (j>=8 ? 16 : 0).
// ---------------------------------------------------------------------------
__global__ void prep_kernel(const float* __restrict__ W_in,   // [64,256]
                            const float* __restrict__ W_h,    // [64,64]
                            _Float16* __restrict__ wfrag1,
                            _Float16* __restrict__ wfrag2,
                            float* __restrict__ gsum,
                            float* __restrict__ gsumsq) {
  const int idx = blockIdx.x * 256 + threadIdx.x;  // 80*256 = 20480 = 16384+4096
  if (idx < WF1_HALFS) {
    const int j    = idx & 15;
    const int lane = (idx >> 4) & 31;
    const int cidx = idx >> 9;            // chunk*4 + nt, 0..31
    const int c    = cidx >> 2;
    const int nt   = cidx & 3;
    const int n    = nt * 16 + (lane & 15);
    const int k    = c * 32 + (lane >> 4) * 8 + (j & 7) + ((j >= 8) ? 16 : 0);
    wfrag1[idx] = (_Float16)W_in[n * 256 + k];
  } else if (idx < WF1_HALFS + WF2_HALFS) {
    const int i2   = idx - WF1_HALFS;
    const int j    = i2 & 15;
    const int lane = (i2 >> 4) & 31;
    const int cidx = i2 >> 9;             // 0..7
    const int c    = cidx >> 2;
    const int nt   = cidx & 3;
    const int n    = nt * 16 + (lane & 15);
    const int k    = c * 32 + (lane >> 4) * 8 + (j & 7) + ((j >= 8) ? 16 : 0);
    wfrag2[i2] = (_Float16)W_h[n * 64 + k];
  }
  if (idx < H_DIM) { gsum[idx] = 0.f; gsumsq[idx] = 0.f; }
}

// Build an A-fragment row-slice: 8 f32 at p, 8 f32 at p+16 -> 16 halfs.
__device__ inline half16 pack_a(const float* __restrict__ p) {
  float4 a0 = *(const float4*)(p + 0);
  float4 a1 = *(const float4*)(p + 4);
  float4 b0 = *(const float4*)(p + 16);
  float4 b1 = *(const float4*)(p + 20);
  half16 r;
  r[0]  = (_Float16)a0.x; r[1]  = (_Float16)a0.y; r[2]  = (_Float16)a0.z; r[3]  = (_Float16)a0.w;
  r[4]  = (_Float16)a1.x; r[5]  = (_Float16)a1.y; r[6]  = (_Float16)a1.z; r[7]  = (_Float16)a1.w;
  r[8]  = (_Float16)b0.x; r[9]  = (_Float16)b0.y; r[10] = (_Float16)b0.z; r[11] = (_Float16)b0.w;
  r[12] = (_Float16)b1.x; r[13] = (_Float16)b1.y; r[14] = (_Float16)b1.z; r[15] = (_Float16)b1.w;
  return r;
}

// ---------------------------------------------------------------------------
// Main kernel (persistent blocks): each wave grid-strides over 16-edge tiles.
//   h1 = relu([x[src]|x[dst]] @ W_in^T + b_in)   via 8x4 v_wmma_f32_16x16x32_f16
//   h2 = h1 @ W_h^T + b_h                        via 2x4 v_wmma_f32_16x16x32_f16
// h2 stored to ws (f32). BN sum/sumsq accumulated in per-lane registers across
// the whole loop, flushed once per wave (LDS) and once per block (global).
// ---------------------------------------------------------------------------
__global__ __launch_bounds__(256) void edge_mlp_kernel(
    const float* __restrict__ x,          // [N_NODES,128]
    const int* __restrict__ edges,        // [2,E]
    const float* __restrict__ b_in,       // [64]
    const float* __restrict__ b_h,        // [64]
    const _Float16* __restrict__ wfrag1,
    const _Float16* __restrict__ wfrag2,
    float* __restrict__ h2buf,            // [E,64]
    float* __restrict__ gsum,             // [64]
    float* __restrict__ gsumsq,           // [64]
    int E, int numTiles) {
  __shared__ alignas(32) _Float16 wf1[WF1_HALFS];
  __shared__ alignas(32) _Float16 wf2[WF2_HALFS];
  __shared__ alignas(32) _Float16 h1s[WAVES_PER_BLOCK][16 * 64];
  __shared__ float bsum[H_DIM];
  __shared__ float bsumsq[H_DIM];

  const int tid = threadIdx.x;
  {  // cooperative weight-fragment load into LDS (once per persistent block)
    const uint32_t* s1 = (const uint32_t*)wfrag1;
    uint32_t*       d1 = (uint32_t*)wf1;
    for (int i = tid; i < WF1_HALFS / 2; i += 256) d1[i] = s1[i];
    const uint32_t* s2 = (const uint32_t*)wfrag2;
    uint32_t*       d2 = (uint32_t*)wf2;
    for (int i = tid; i < WF2_HALFS / 2; i += 256) d2[i] = s2[i];
    if (tid < H_DIM) { bsum[tid] = 0.f; bsumsq[tid] = 0.f; }
  }
  __syncthreads();

  const int wave = tid >> 5;
  const int lane = tid & 31;
  const int nlo  = lane & 15;
  const int hi   = lane >> 4;
  const int tileStride = gridDim.x * WAVES_PER_BLOCK;

  // Per-lane BN stat accumulators: channel n = nt*16 + nlo is fixed per lane.
  float lsum[4]   = {0.f, 0.f, 0.f, 0.f};
  float lsumsq[4] = {0.f, 0.f, 0.f, 0.f};

  // Wave-uniform loop bound: EXEC stays all-ones for WMMA.
  for (int tile = blockIdx.x * WAVES_PER_BLOCK + wave; tile < numTiles;
       tile += tileStride) {
    const int e0   = tile * 16;
    const int myE  = e0 + nlo;
    const int srcN = edges[myE];
    const int dstN = edges[E + myE];
    const float* srcRow = x + (size_t)srcN * F_DIM;
    const float* dstRow = x + (size_t)dstN * F_DIM;

    // ---- GEMM1: [16 x 256] x [256 x 64] ----
    float8 acc[4] = {};
#pragma unroll
    for (int c = 0; c < 8; ++c) {
      const float* row = (c < 4) ? srcRow : dstRow;
      const int f0 = (c & 3) * 32 + hi * 8;
      half16 a = pack_a(row + f0);
#pragma unroll
      for (int nt = 0; nt < 4; ++nt) {
        half16 b = *(const half16*)&wf1[((c * 4 + nt) * 32 + lane) * 16];
        acc[nt] = __builtin_amdgcn_wmma_f32_16x16x32_f16(
            false, a, false, b, (short)0, acc[nt], false, false);
      }
    }

    // ---- bias + relu, stage h1 into per-wave LDS tile [16][64] f16 ----
    _Float16* hbase = &h1s[wave][0];
#pragma unroll
    for (int nt = 0; nt < 4; ++nt) {
      const int n  = nt * 16 + nlo;
      const float bi = b_in[n];
#pragma unroll
      for (int r = 0; r < 8; ++r) {
        const int m = hi * 8 + r;
        float v = acc[nt][r] + bi;
        v = v > 0.f ? v : 0.f;
        hbase[m * 64 + n] = (_Float16)v;
      }
    }
    // LDS ops are in-order within a wave; wait for stores before re-reading
    asm volatile("s_wait_dscnt 0" ::: "memory");

    // ---- GEMM2: [16 x 64] x [64 x 64] ----
    float8 acc2[4] = {};
    const _Float16* hrow = &h1s[wave][nlo * 64];
#pragma unroll
    for (int c = 0; c < 2; ++c) {
      const int base = c * 32 + hi * 8;
      half8 p0 = *(const half8*)(hrow + base);
      half8 p1 = *(const half8*)(hrow + base + 16);
      half16 a2 = __builtin_shufflevector(p0, p1, 0, 1, 2, 3, 4, 5, 6, 7,
                                          8, 9, 10, 11, 12, 13, 14, 15);
#pragma unroll
      for (int nt = 0; nt < 4; ++nt) {
        half16 b2 = *(const half16*)&wf2[((c * 4 + nt) * 32 + lane) * 16];
        acc2[nt] = __builtin_amdgcn_wmma_f32_16x16x32_f16(
            false, a2, false, b2, (short)0, acc2[nt], false, false);
      }
    }

    // ---- bias, store h2, accumulate BN partial stats in registers ----
#pragma unroll
    for (int nt = 0; nt < 4; ++nt) {
      const int n  = nt * 16 + nlo;
      const float bh = b_h[n];
#pragma unroll
      for (int r = 0; r < 8; ++r) {
        const int m = hi * 8 + r;
        float v = acc2[nt][r] + bh;
        h2buf[(size_t)(e0 + m) * H_DIM + n] = v;
        lsum[nt] += v;
        lsumsq[nt] += v * v;
      }
    }
  }

  // Flush per-lane stats: LDS atomics once per wave, then global once per block.
#pragma unroll
  for (int nt = 0; nt < 4; ++nt) {
    const int n = nt * 16 + nlo;
    atomicAdd(&bsum[n], lsum[nt]);
    atomicAdd(&bsumsq[n], lsumsq[nt]);
  }
  __syncthreads();
  if (tid < H_DIM) {
    atomicAdd(&gsum[tid], bsum[tid]);
    atomicAdd(&gsumsq[tid], bsumsq[tid]);
  }
}

// ---------------------------------------------------------------------------
// Fold BN stats + gamma/beta into per-channel scale/shift.
// ---------------------------------------------------------------------------
__global__ void bn_finalize_kernel(const float* __restrict__ gsum,
                                   const float* __restrict__ gsumsq,
                                   const float* __restrict__ gamma,
                                   const float* __restrict__ beta,
                                   float* __restrict__ scale,
                                   float* __restrict__ shift, float invE) {
  const int n = threadIdx.x;
  if (n < H_DIM) {
    const float mean = gsum[n] * invE;
    const float var  = gsumsq[n] * invE - mean * mean;
    const float inv  = rsqrtf(var + BN_EPS);
    const float sc   = gamma[n] * inv;
    scale[n] = sc;
    shift[n] = beta[n] - mean * sc;
  }
}

// ---------------------------------------------------------------------------
// Pass 2: out[e] = b_out + sum_n relu(h2[e,n]*scale[n]+shift[n]) * W_out[n]
// ---------------------------------------------------------------------------
__global__ __launch_bounds__(256) void score_kernel(
    const float* __restrict__ h2buf, const float* __restrict__ scale,
    const float* __restrict__ shift, const float* __restrict__ wout,
    const float* __restrict__ bout, float* __restrict__ out, int E) {
  __shared__ float s_sc[H_DIM], s_sh[H_DIM], s_w[H_DIM];
  const int tid = threadIdx.x;
  if (tid < H_DIM) { s_sc[tid] = scale[tid]; s_sh[tid] = shift[tid]; s_w[tid] = wout[tid]; }
  __syncthreads();
  const int e = blockIdx.x * 256 + tid;
  if (e >= E) return;
  const float4* hp = (const float4*)(h2buf + (size_t)e * H_DIM);
  float s = bout[0];
#pragma unroll
  for (int i = 0; i < 16; ++i) {
    float4 v = hp[i];
    const int n = i * 4;
    s += fmaxf(v.x * s_sc[n + 0] + s_sh[n + 0], 0.f) * s_w[n + 0];
    s += fmaxf(v.y * s_sc[n + 1] + s_sh[n + 1], 0.f) * s_w[n + 1];
    s += fmaxf(v.z * s_sc[n + 2] + s_sh[n + 2], 0.f) * s_w[n + 2];
    s += fmaxf(v.w * s_sc[n + 3] + s_sh[n + 3], 0.f) * s_w[n + 3];
  }
  out[e] = s;
}

extern "C" void kernel_launch(void* const* d_in, const int* in_sizes, int n_in,
                              void* d_out, int out_size, void* d_ws, size_t ws_size,
                              hipStream_t stream) {
  const float* x     = (const float*)d_in[0];
  const int*   edges = (const int*)d_in[1];
  const float* W_in  = (const float*)d_in[2];
  const float* b_in  = (const float*)d_in[3];
  const float* W_h   = (const float*)d_in[4];
  const float* b_h   = (const float*)d_in[5];
  const float* gamma = (const float*)d_in[6];
  const float* beta  = (const float*)d_in[7];
  const float* W_out = (const float*)d_in[8];
  const float* b_out = (const float*)d_in[9];
  const int E = in_sizes[1] / 2;

  // Workspace layout
  char*  ws     = (char*)d_ws;
  float* h2buf  = (float*)ws;
  size_t off    = (size_t)E * H_DIM * sizeof(float);
  float* gsum   = (float*)(ws + off); off += H_DIM * sizeof(float);
  float* gsumsq = (float*)(ws + off); off += H_DIM * sizeof(float);
  float* scale  = (float*)(ws + off); off += H_DIM * sizeof(float);
  float* shift  = (float*)(ws + off); off += H_DIM * sizeof(float);
  _Float16* wfrag1 = (_Float16*)(ws + off); off += WF1_HALFS * sizeof(_Float16);
  _Float16* wfrag2 = (_Float16*)(ws + off); off += WF2_HALFS * sizeof(_Float16);

  prep_kernel<<<(WF1_HALFS + WF2_HALFS) / 256, 256, 0, stream>>>(
      W_in, W_h, wfrag1, wfrag2, gsum, gsumsq);

  const int numTiles = E / 16;
  int blocksA = (numTiles + WAVES_PER_BLOCK - 1) / WAVES_PER_BLOCK;
  if (blocksA > PERSISTENT_BLOCKS) blocksA = PERSISTENT_BLOCKS;
  edge_mlp_kernel<<<blocksA, 256, 0, stream>>>(
      x, edges, b_in, b_h, wfrag1, wfrag2, h2buf, gsum, gsumsq, E, numTiles);

  bn_finalize_kernel<<<1, 64, 0, stream>>>(gsum, gsumsq, gamma, beta, scale,
                                           shift, 1.0f / (float)E);

  score_kernel<<<(E + 255) / 256, 256, 0, stream>>>(h2buf, scale, shift, W_out,
                                                    b_out, (float*)d_out, E);
}
